// GNN_57329223467619
// MI455X (gfx1250) — matrix operand order repaired and checked by reading, not compiled, not run on hardware
//
#include <hip/hip_runtime.h>
#include <hip/hip_bf16.h>
#include <stdint.h>

typedef __bf16 bf16;
typedef __bf16 v16bf __attribute__((ext_vector_type(16)));
typedef __bf16 v8bf  __attribute__((ext_vector_type(8)));
typedef float  v8f   __attribute__((ext_vector_type(8)));
typedef unsigned int v4u __attribute__((ext_vector_type(4)));
typedef int v8i __attribute__((ext_vector_type(8)));
typedef int v4i __attribute__((ext_vector_type(4)));

#define BM 128
#define BN 128
#define BK 32

#if defined(__HIP_DEVICE_COMPILE__) && __has_builtin(__builtin_amdgcn_tensor_load_to_lds) && __has_builtin(__builtin_amdgcn_s_wait_tensorcnt)
#define USE_TDM 1
#else
#define USE_TDM 0
#endif

#if defined(__HIP_DEVICE_COMPILE__) && __has_builtin(__builtin_amdgcn_sched_barrier)
#define SCHED_FENCE() __builtin_amdgcn_sched_barrier(0)
#else
#define SCHED_FENCE()
#endif

// ---------------------------------------------------------------------------
// Prep: one block per adjacency row. Counts nonzeros (degree), converts the
// raw 0/1 row to bf16 (exact), writes d[row] = rsqrt(deg).
// ---------------------------------------------------------------------------
__global__ __launch_bounds__(256) void k_prep_adj(const float* __restrict__ adj,
                                                  bf16* __restrict__ Ab,
                                                  float* __restrict__ d, int n)
{
    __shared__ int cnt[256];
    const int row = blockIdx.x;
    const float4* src4 = (const float4*)(adj + (long)row * n);
    bf16* dst = Ab + (long)row * n;
    int c = 0;
    for (int j = threadIdx.x; j < n / 4; j += 256) {
        float4 v = src4[j];
        c += (v.x != 0.0f) + (v.y != 0.0f) + (v.z != 0.0f) + (v.w != 0.0f);
        union { unsigned long long u; bf16 h[4]; } p;
        p.h[0] = (bf16)v.x; p.h[1] = (bf16)v.y; p.h[2] = (bf16)v.z; p.h[3] = (bf16)v.w;
        *(unsigned long long*)(dst + 4 * j) = p.u;
    }
    cnt[threadIdx.x] = c;
    __syncthreads();
    for (int s = 128; s > 0; s >>= 1) {
        if (threadIdx.x < s) cnt[threadIdx.x] += cnt[threadIdx.x + s];
        __syncthreads();
    }
    if (threadIdx.x == 0) d[row] = rsqrtf((float)cnt[0]);
}

// ---------------------------------------------------------------------------
// f32 -> bf16 with zero padding to (drows, dcols)
// ---------------------------------------------------------------------------
__global__ void k_cvt_pad(const float* __restrict__ src, bf16* __restrict__ dst,
                          int srows, int scols, int drows, int dcols)
{
    long i = (long)blockIdx.x * 256 + threadIdx.x;
    if (i >= (long)drows * dcols) return;
    int r = (int)(i / dcols), c = (int)(i % dcols);
    dst[i] = (r < srows && c < scols) ? (bf16)src[(long)r * scols + c] : (bf16)0.0f;
}

__global__ void k_zero(float* __restrict__ p, int n)
{
    int i = blockIdx.x * 256 + threadIdx.x;
    if (i < n) p[i] = 0.0f;
}

// ---------------------------------------------------------------------------
// TDM: issue a 2D tile load (tile_d1 rows x tile_d0 cols, 2-byte elements)
// from a (tensor_d1 x tensor_d0) row-major bf16 tensor into LDS.
// OOB rows/cols (beyond tensor dims) are written as zero by the TDM.
// Must be reached by exactly ONE wave (TDM ignores EXEC; caller branches on a
// uniform SGPR predicate). 6-arg builtin: (v4u, v8i, v4i, v4i, v8i, cpol).
// ---------------------------------------------------------------------------
#if USE_TDM
__device__ inline void tdm_load_tile(const void* gsrc, const void* lds_dst,
                                     unsigned tensor_d0, unsigned tensor_d1,
                                     unsigned tile_d0, unsigned tile_d1,
                                     unsigned stride0)
{
    unsigned long long ga = (unsigned long long)(uintptr_t)gsrc;
    v4u g0;
    g0[0] = 1u;                                        // count=1 (valid user D#)
    g0[1] = (unsigned)(uintptr_t)lds_dst;              // lds_addr (bits 63:32)
    g0[2] = (unsigned)ga;                              // global_addr[31:0]
    g0[3] = ((unsigned)(ga >> 32) & 0x01ffffffu)       // global_addr[56:32]
          | (2u << 30);                                // type=2 ("image")
    v8i g1;
    g1[0] = (int)(1u << 16);                           // data_size=1 -> 2 bytes
    g1[1] = (int)((tensor_d0 & 0xffffu) << 16);        // tensor_dim0[15:0]
    g1[2] = (int)((tensor_d0 >> 16) | ((tensor_d1 & 0xffffu) << 16)); // dim0 hi | dim1 lo
    g1[3] = (int)((tensor_d1 >> 16) | (tile_d0 << 16));               // dim1 hi | tile_dim0
    g1[4] = (int)(tile_d1 & 0xffffu);                  // tile_dim1 | tile_dim2=0
    g1[5] = (int)stride0;                              // tensor_dim0_stride[31:0]
    g1[6] = 0;                                         // stride0 hi | dim1_stride lo
    g1[7] = 0;
    v4i g2 = (v4i)(0);                                 // group2: unused dims (2D tile)
    v4i g3 = (v4i)(0);                                 // group3: unused
    v8i g4 = (v8i)(0);                                 // trailing descriptor words: unused
    __builtin_amdgcn_tensor_load_to_lds(g0, g1, g2, g3, g4, 0);
}
#endif

// ---------------------------------------------------------------------------
// Split staging: global->regs (issue early, overlap with WMMA) and
// regs->LDS (issue after WMMA; its loadcnt wait is off the critical path).
// B tile (BK x BN) transposed into Bs[n][k]; A tile (BM x BK) row-major.
// ---------------------------------------------------------------------------
__device__ inline void stage_b_load(const bf16* __restrict__ B, unsigned breg[8],
                                    int k0, int n0, int N, int K, int ldb, int tid)
{
#pragma unroll
    for (int j = 0; j < 8; ++j) {
        int i = tid + j * 256;
        int k = i >> 6;               // 64 dword-chunks per k-row
        int n = (i & 63) * 2;
        int gk = k0 + k, gn = n0 + n;
        unsigned u = 0;
        if (gk < K && gn + 1 < N)
            u = *(const unsigned int*)(B + (long)gk * ldb + gn);
        breg[j] = u;
    }
}

__device__ inline void stage_b_store(bf16* Bs, const unsigned breg[8], int tid)
{
#pragma unroll
    for (int j = 0; j < 8; ++j) {
        int i = tid + j * 256;
        int k = i >> 6;
        int n = (i & 63) * 2;
        union { unsigned int u; bf16 h[2]; } p;
        p.u = breg[j];
        Bs[(n + 0) * BK + k] = p.h[0];
        Bs[(n + 1) * BK + k] = p.h[1];
    }
}

__device__ inline void stage_a_load(const bf16* __restrict__ A, unsigned areg[8],
                                    int k0, int m0, int M, int K, int lda, int tid)
{
#pragma unroll
    for (int j = 0; j < 8; ++j) {
        int i = tid + j * 256;
        int r = i >> 4;               // 16 dword-chunks per row
        int c = (i & 15) * 2;
        int gm = m0 + r, gk = k0 + c;
        unsigned u = 0;
        if (gm < M && gk + 1 < K)
            u = *(const unsigned int*)(A + (long)gm * lda + gk);
        areg[j] = u;
    }
}

__device__ inline void stage_a_store(bf16* As, const unsigned areg[8], int tid)
{
#pragma unroll
    for (int j = 0; j < 8; ++j) {
        int i = tid + j * 256;
        int r = i >> 4;
        int c = (i & 15) * 2;
        *(unsigned int*)(&As[r * BK + c]) = areg[j];
    }
}

__device__ inline void compute_tile(const bf16* As, const bf16* Bs,
                                    v8f acc[8], int wave, int lane)
{
    const int row = (wave << 4) + (lane & 15);
    const int kb  = (lane >> 4) << 4;          // 0 or 16
    v8bf lo = *(const v8bf*)(&As[row * BK + kb]);
    v8bf hi = *(const v8bf*)(&As[row * BK + kb + 8]);
    v16bf afrag = __builtin_shufflevector(lo, hi,
        0, 1, 2, 3, 4, 5, 6, 7, 8, 9, 10, 11, 12, 13, 14, 15);
    const int kb2 = (lane >> 4) << 4;          // 0 or 16
    v16bf bfr[8];
#pragma unroll
    for (int j = 0; j < 8; ++j)
        bfr[j] = *(const v16bf*)(&Bs[((j << 4) + (lane & 15)) * BK + kb2]);
#pragma unroll
    for (int j = 0; j < 8; ++j)
        acc[j] = __builtin_amdgcn_wmma_f32_16x16x32_bf16(
            false, afrag, false, bfr[j], (short)0, acc[j], false, false);
}

// ---------------------------------------------------------------------------
// WMMA GEMM (non-transposed A): C = rowscale[m]*(A@B)[m,n] + bias[n]
// A [M x K] row-major (lda), B [K x N] row-major (ldb).  REQUIRES K % 64 == 0.
// Ping-pong LDS buffers, unrolled x2. A tiles via Tensor Data Mover (wave 0
// only; scalar branch), B tiles via global->reg loads issued before the WMMA
// block and reg->LDS stores fenced after it (sched_barrier), so the WMMA
// burst never waits on global latency.
// ---------------------------------------------------------------------------
__global__ __launch_bounds__(256) void k_wmma_gemm_nn(
    const bf16* __restrict__ A, const bf16* __restrict__ B,
    float* __restrict__ C, bf16* __restrict__ Cb,
    const float* __restrict__ rowscale, const float* __restrict__ bias,
    int M, int N, int K, int lda, int ldb, int ldc)
{
    __shared__ __align__(32) bf16 As[2][BM * BK];
    __shared__ __align__(32) bf16 Bs[2][BN * BK];

    const int tid  = threadIdx.x;
    const int lane = tid & 31;
    const int wave = tid >> 5;
    // uniform (SGPR) wave-0 predicate -> real scalar branch around TDM ops
    const bool wave0 = (__builtin_amdgcn_readfirstlane(tid) < 32);
    const int m0 = blockIdx.y * BM;
    const int n0 = blockIdx.x * BN;

    v8f acc[8];
#pragma unroll
    for (int j = 0; j < 8; ++j)
#pragma unroll
        for (int e = 0; e < 8; ++e) acc[j][e] = 0.0f;

    // prologue: fetch tile 0 into buffer 0
#if USE_TDM
    if (wave0)
        tdm_load_tile(A + (size_t)m0 * lda, &As[0][0],
                      (unsigned)lda, (unsigned)M, BK, BM, (unsigned)lda);
#else
    {
        unsigned areg[8];
        stage_a_load(A, areg, 0, m0, M, K, lda, tid);
        stage_a_store(&As[0][0], areg, tid);
    }
#endif
    {
        unsigned breg[8];
        stage_b_load(B, breg, 0, n0, N, K, ldb, tid);
        stage_b_store(&Bs[0][0], breg, tid);
    }

    for (int k0 = 0; k0 < K; k0 += 2 * BK) {
        // ---- phase 0: consume buffer 0, prefetch k0+BK into buffer 1 ----
#if USE_TDM
        if (wave0) __builtin_amdgcn_s_wait_tensorcnt(0);
#endif
        __syncthreads();
        unsigned breg0[8];
        stage_b_load(B, breg0, k0 + BK, n0, N, K, ldb, tid);  // k0+BK<K always
#if USE_TDM
        if (wave0)
            tdm_load_tile(A + (size_t)m0 * lda + (k0 + BK), &As[1][0],
                          (unsigned)lda, (unsigned)M, BK, BM, (unsigned)lda);
#else
        unsigned areg0[8];
        stage_a_load(A, areg0, k0 + BK, m0, M, K, lda, tid);
#endif
        compute_tile(&As[0][0], &Bs[0][0], acc, wave, lane);
        SCHED_FENCE();   // keep deferred stores (and their loadcnt wait) below the WMMA burst
        stage_b_store(&Bs[1][0], breg0, tid);
#if !USE_TDM
        stage_a_store(&As[1][0], areg0, tid);
#endif

        // ---- phase 1: consume buffer 1, prefetch k0+2*BK into buffer 0 ----
#if USE_TDM
        if (wave0) __builtin_amdgcn_s_wait_tensorcnt(0);
#endif
        __syncthreads();
        const bool pf = (k0 + 2 * BK < K);
        unsigned breg1[8];
        if (pf) stage_b_load(B, breg1, k0 + 2 * BK, n0, N, K, ldb, tid);
#if USE_TDM
        if (wave0 && pf)
            tdm_load_tile(A + (size_t)m0 * lda + (k0 + 2 * BK), &As[0][0],
                          (unsigned)lda, (unsigned)M, BK, BM, (unsigned)lda);
#else
        unsigned areg1[8];
        if (pf) stage_a_load(A, areg1, k0 + 2 * BK, m0, M, K, lda, tid);
#endif
        compute_tile(&As[1][0], &Bs[1][0], acc, wave, lane);
        SCHED_FENCE();
        if (pf) {
            stage_b_store(&Bs[0][0], breg1, tid);
#if !USE_TDM
            stage_a_store(&As[0][0], areg1, tid);
#endif
        }
    }

    // epilogue: lanes 0-15 -> M=r, lanes 16-31 -> M=8+r; N = lane&15 within group
    const int rr = (lane >> 4) << 3;
    const int c0 = lane & 15;
#pragma unroll
    for (int j = 0; j < 8; ++j) {
        const int col = n0 + (j << 4) + c0;
        if (col >= N) continue;
        const float bv = bias ? bias[col] : 0.0f;
#pragma unroll
        for (int r = 0; r < 8; ++r) {
            const int row = m0 + (wave << 4) + rr + r;
            if (row >= M) continue;
            float v = acc[j][r];
            if (rowscale) v *= rowscale[row];
            v += bv;
            const long idx = (long)row * ldc + col;
            if (C)  C[idx]  = v;
            if (Cb) Cb[idx] = (bf16)v;
        }
    }
}

// ---------------------------------------------------------------------------
// WMMA GEMM, A transposed, split-K: C += (A^T @ B) over this block's K-chunk.
// A stored [K x M] row-major. C must be pre-zeroed; accumulation via native
// f32 global atomics. Grid z = K / kchunk. Used for s1^T @ {P, z1}.
// ---------------------------------------------------------------------------
__global__ __launch_bounds__(256) void k_wmma_gemm_tn(
    const bf16* __restrict__ A, const bf16* __restrict__ B,
    float* __restrict__ C,
    int M, int N, int K, int lda, int ldb, int ldc, int kchunk)
{
    __shared__ __align__(32) bf16 As[BM * BK];
    __shared__ __align__(32) bf16 Bs[BN * BK];

    const int tid  = threadIdx.x;
    const int lane = tid & 31;
    const int wave = tid >> 5;
    const int m0 = blockIdx.y * BM;
    const int n0 = blockIdx.x * BN;
    const int kbeg = blockIdx.z * kchunk;
    const int kend = (kbeg + kchunk < K) ? (kbeg + kchunk) : K;

    v8f acc[8];
#pragma unroll
    for (int j = 0; j < 8; ++j)
#pragma unroll
        for (int e = 0; e < 8; ++e) acc[j][e] = 0.0f;

    for (int k0 = kbeg; k0 < kend; k0 += BK) {
        // A tile: A[k, m] contiguous in m -> dword loads, transposed store
#pragma unroll
        for (int i = tid; i < (BM * BK) / 2; i += 256) {
            int k = i >> 6;
            int m = (i & 63) * 2;
            int gk = k0 + k, gm = m0 + m;
            union { unsigned int u; bf16 h[2]; } p;
            p.u = 0u;
            if (gk < K) {
                if (gm + 1 < M)  p.u = *(const unsigned int*)(A + (long)gk * lda + gm);
                else if (gm < M) p.h[0] = A[(long)gk * lda + gm];
            }
            As[(m + 0) * BK + k] = p.h[0];
            As[(m + 1) * BK + k] = p.h[1];
        }
        {
            unsigned breg[8];
            stage_b_load(B, breg, k0, n0, N, K, ldb, tid);
            stage_b_store(Bs, breg, tid);
        }
        __syncthreads();
        compute_tile(As, Bs, acc, wave, lane);
        __syncthreads();
    }

    const int rr = (lane >> 4) << 3;
    const int c0 = lane & 15;
#pragma unroll
    for (int j = 0; j < 8; ++j) {
        const int col = n0 + (j << 4) + c0;
        if (col >= N) continue;
#pragma unroll
        for (int r = 0; r < 8; ++r) {
            const int row = m0 + (wave << 4) + rr + r;
            if (row >= M) continue;
            unsafeAtomicAdd(&C[(long)row * ldc + col], acc[j][r]);
        }
    }
}

// ---------------------------------------------------------------------------
// Stage-1 softmax over 256-wide rows; emits s1 (bf16) and d-prescaled s1
// ---------------------------------------------------------------------------
__global__ __launch_bounds__(256) void k_softmax_s1(const float* __restrict__ pre,
                                                    const float* __restrict__ d,
                                                    bf16* __restrict__ s,
                                                    bf16* __restrict__ sd, int cols)
{
    __shared__ float red[256];
    const int row = blockIdx.x;
    const int t = threadIdx.x;
    const float* x = pre + (long)row * cols;
    float v = (t < cols) ? x[t] : -3.4e38f;
    red[t] = v; __syncthreads();
    for (int st = 128; st > 0; st >>= 1) {
        if (t < st) red[t] = fmaxf(red[t], red[t + st]);
        __syncthreads();
    }
    float mx = red[0]; __syncthreads();
    float e = (t < cols) ? expf(v - mx) : 0.0f;
    red[t] = e; __syncthreads();
    for (int st = 128; st > 0; st >>= 1) {
        if (t < st) red[t] += red[t + st];
        __syncthreads();
    }
    float inv = 1.0f / red[0];
    if (t < cols) {
        float p = e * inv;
        s [(long)row * cols + t] = (bf16)p;
        sd[(long)row * cols + t] = (bf16)(p * d[row]);
    }
}

// ---------------------------------------------------------------------------
// Small f32 GEMM (post-pool stages; negligible FLOPs). transA: A stored [K x M].
// ---------------------------------------------------------------------------
__global__ void k_sgemm(const float* __restrict__ A, const float* __restrict__ B,
                        const float* __restrict__ bias, float* __restrict__ C,
                        int M, int N, int K, int transA)
{
    int idx = blockIdx.x * blockDim.x + threadIdx.x;
    if (idx >= M * N) return;
    int m = idx / N, n = idx % N;
    float s = 0.0f;
    if (transA) { for (int k = 0; k < K; ++k) s += A[k * M + m] * B[k * N + n]; }
    else        { for (int k = 0; k < K; ++k) s += A[m * K + k] * B[k * N + n]; }
    C[idx] = s + (bias ? bias[n] : 0.0f);
}

// In-place row softmax, cols <= 256, one block per row
__global__ __launch_bounds__(256) void k_softmax_rows(float* __restrict__ x, int cols)
{
    __shared__ float red[256];
    const int row = blockIdx.x;
    const int t = threadIdx.x;
    float* p = x + (long)row * cols;
    float v = (t < cols) ? p[t] : -3.4e38f;
    red[t] = v; __syncthreads();
    for (int st = 128; st > 0; st >>= 1) {
        if (t < st) red[t] = fmaxf(red[t], red[t + st]);
        __syncthreads();
    }
    float mx = red[0]; __syncthreads();
    float e = (t < cols) ? expf(v - mx) : 0.0f;
    red[t] = e; __syncthreads();
    for (int st = 128; st > 0; st >>= 1) {
        if (t < st) red[t] += red[t + st];
        __syncthreads();
    }
    if (t < cols) p[t] = e / red[0];
}

// Final: out = log_softmax(xf @ Wfc + bfc), xf [1x128], Wfc [128x10]
__global__ __launch_bounds__(64) void k_final(const float* __restrict__ xf,
                                              const float* __restrict__ Wfc,
                                              const float* __restrict__ bfc,
                                              float* __restrict__ out)
{
    __shared__ float logits[10];
    const int t = threadIdx.x;
    if (t < 10) {
        float s = bfc[t];
        for (int k = 0; k < 128; ++k) s += xf[k] * Wfc[k * 10 + t];
        logits[t] = s;
    }
    __syncthreads();
    if (t == 0) {
        float mx = -3.4e38f;
        for (int i = 0; i < 10; ++i) mx = fmaxf(mx, logits[i]);
        float sum = 0.0f;
        for (int i = 0; i < 10; ++i) sum += expf(logits[i] - mx);
        float lse = mx + logf(sum);
        for (int i = 0; i < 10; ++i) out[i] = logits[i] - lse;
    }
}

// ---------------------------------------------------------------------------
extern "C" void kernel_launch(void* const* d_in, const int* in_sizes, int n_in,
                              void* d_out, int out_size, void* d_ws, size_t ws_size,
                              hipStream_t stream)
{
    (void)in_sizes; (void)n_in; (void)out_size; (void)ws_size;
    const float* x    = (const float*)d_in[0];
    const float* adj  = (const float*)d_in[1];
    const float* W1   = (const float*)d_in[2];  const float* b1   = (const float*)d_in[3];
    const float* Wd11 = (const float*)d_in[4];  const float* bd11 = (const float*)d_in[5];
    const float* Wd12 = (const float*)d_in[6];  const float* bd12 = (const float*)d_in[7];
    const float* W2   = (const float*)d_in[8];  const float* b2   = (const float*)d_in[9];
    const float* Wd21 = (const float*)d_in[10]; const float* bd21 = (const float*)d_in[11];
    const float* Wd22 = (const float*)d_in[12]; const float* bd22 = (const float*)d_in[13];
    const float* W3   = (const float*)d_in[14]; const float* b3   = (const float*)d_in[15];
    const float* Wd31 = (const float*)d_in[16]; const float* bd31 = (const float*)d_in[17];
    const float* Wd32 = (const float*)d_in[18]; const float* bd32 = (const float*)d_in[19];
    const float* Wfc  = (const float*)d_in[20]; const float* bfc  = (const float*)d_in[21];
    float* out = (float*)d_out;

    const int N = 8192;
    const int KF = 832;                // 784 padded up to a multiple of 64
    char* w = (char*)d_ws;
    size_t off = 0;
    auto take = [&](size_t bytes) -> char* {
        char* p = w + off;
        off = (off + bytes + 255) & ~(size_t)255;
        return p;
    };
    bf16*  Ab    = (bf16*) take((size_t)N * N * 2);          // 128 MB bf16 adjacency (L2-resident)
    float* dv    = (float*)take((size_t)N * 4);
    bf16*  xb    = (bf16*) take((size_t)N * KF * 2);         // x padded K 784->832
    bf16*  Wb1   = (bf16*) take((size_t)KF * 256 * 2);
    bf16*  Wb11  = (bf16*) take((size_t)256 * 128 * 2);
    bf16*  Wb12  = (bf16*) take((size_t)256 * 256 * 2);
    bf16*  T0b   = (bf16*) take((size_t)N * 256 * 2);
    bf16*  X1b   = (bf16*) take((size_t)N * 256 * 2);
    bf16*  U1b   = (bf16*) take((size_t)N * 128 * 2);
    bf16*  U2b   = (bf16*) take((size_t)N * 256 * 2);
    bf16*  z1b   = (bf16*) take((size_t)N * 128 * 2);
    float* s1pre = (float*)take((size_t)N * 256 * 4);
    bf16*  s1b   = (bf16*) take((size_t)N * 256 * 2);
    bf16*  s1db  = (bf16*) take((size_t)N * 256 * 2);
    bf16*  Pb    = (bf16*) take((size_t)N * 256 * 2);
    float* adj2  = (float*)take((size_t)256 * 256 * 4);
    float* x2    = (float*)take((size_t)256 * 128 * 4);
    float* tmpA  = (float*)take((size_t)256 * 128 * 4);
    float* xg2   = (float*)take((size_t)256 * 128 * 4);
    float* z2    = (float*)take((size_t)256 * 128 * 4);
    float* s2    = (float*)take((size_t)256 * 64 * 4);
    float* x3    = (float*)take((size_t)64 * 128 * 4);
    float* Q     = (float*)take((size_t)256 * 64 * 4);
    float* adj3  = (float*)take((size_t)64 * 64 * 4);
    float* xg3   = (float*)take((size_t)64 * 128 * 4);
    float* z3    = (float*)take((size_t)64 * 128 * 4);
    float* s3    = (float*)take((size_t)64 * 4);
    float* xf    = (float*)take((size_t)128 * 4);

    // degree + bf16 adjacency + d = rsqrt(deg)
    k_prep_adj<<<N, 256, 0, stream>>>(adj, Ab, dv, N);

    auto cvt = [&](const float* s, bf16* dst, int sr, int sc, int dr, int dc) {
        long tot = (long)dr * dc;
        k_cvt_pad<<<(unsigned)((tot + 255) / 256), 256, 0, stream>>>(s, dst, sr, sc, dr, dc);
    };
    cvt(x,    xb,   N,   784, N,   KF);
    cvt(W1,   Wb1,  784, 256, KF,  256);
    cvt(Wd11, Wb11, 256, 128, 256, 128);
    cvt(Wd12, Wb12, 256, 256, 256, 256);

    auto gemm_nn = [&](const bf16* A, const bf16* B, float* C, bf16* Cb,
                       const float* rs, const float* bias,
                       int M, int Nn, int K, int lda, int ldb, int ldc) {
        dim3 g((Nn + BN - 1) / BN, (M + BM - 1) / BM);
        k_wmma_gemm_nn<<<g, 256, 0, stream>>>(A, B, C, Cb, rs, bias, M, Nn, K, lda, ldb, ldc);
    };
    auto gemm_tn = [&](const bf16* A, const bf16* B, float* C,
                       int M, int Nn, int K, int lda, int ldb, int ldc) {
        const int kchunk = 512;
        dim3 g((Nn + BN - 1) / BN, (M + BM - 1) / BM, K / kchunk);
        k_wmma_gemm_tn<<<g, 256, 0, stream>>>(A, B, C, M, Nn, K, lda, ldb, ldc, kchunk);
    };

    // layer 1:  adjn @ Y = d*(A @ (d*Y)); d folded into GEMM epilogues
    gemm_nn(xb,  Wb1,  nullptr, T0b, dv, nullptr, N, 256, KF,  KF,  256, 256); // T0 = d*(x@W1)
    gemm_nn(Ab,  T0b,  nullptr, X1b, dv, b1,      N, 256, N,   N,   256, 256); // X1 = d*(A@T0)+b1
    gemm_nn(X1b, Wb11, nullptr, U1b, dv, nullptr, N, 128, 256, 256, 128, 128); // U1 = d*(X1@Wd11)
    gemm_nn(X1b, Wb12, nullptr, U2b, dv, nullptr, N, 256, 256, 256, 256, 256); // U2 = d*(X1@Wd12)
    gemm_nn(Ab,  U1b,  nullptr, z1b, dv, bd11,    N, 128, N,   N,   128, 128); // z1 = d*(A@U1)+b
    gemm_nn(Ab,  U2b,  s1pre, nullptr, dv, bd12,  N, 256, N,   N,   256, 256); // s1pre
    k_softmax_s1<<<N, 256, 0, stream>>>(s1pre, dv, s1b, s1db, 256);            // s1, d*s1
    gemm_nn(Ab,  s1db, nullptr, Pb, dv, nullptr,  N, 256, N,   N,   256, 256); // P = adjn@s1
    k_zero<<<(256 * 256 + 255) / 256, 256, 0, stream>>>(adj2, 256 * 256);
    k_zero<<<(256 * 128 + 255) / 256, 256, 0, stream>>>(x2,   256 * 128);
    gemm_tn(s1b, Pb,  adj2, 256, 256, N, 256, 256, 256);                       // adj2 = s1^T@P
    gemm_tn(s1b, z1b, x2,   256, 128, N, 256, 128, 128);                       // x2  = s1^T@z1

    // post-pool stages (tiny)
    auto sg = [&](const float* A, const float* B, const float* bias, float* C,
                  int M, int Nn, int K, int tA) {
        int tot = M * Nn;
        k_sgemm<<<(tot + 255) / 256, 256, 0, stream>>>(A, B, bias, C, M, Nn, K, tA);
    };
    sg(x2,   W2,   nullptr, tmpA, 256, 128, 128, 0);
    sg(adj2, tmpA, b2,      xg2,  256, 128, 256, 0);   // gcn2
    sg(xg2,  Wd21, nullptr, tmpA, 256, 128, 128, 0);
    sg(adj2, tmpA, bd21,    z2,   256, 128, 256, 0);   // z2
    sg(xg2,  Wd22, nullptr, tmpA, 256, 64,  128, 0);
    sg(adj2, tmpA, bd22,    s2,   256, 64,  256, 0);
    k_softmax_rows<<<256, 256, 0, stream>>>(s2, 64);   // s2
    sg(s2,   z2,   nullptr, x3,   64, 128, 256, 1);    // x3 = s2^T@z2
    sg(adj2, s2,   nullptr, Q,    256, 64, 256, 0);
    sg(s2,   Q,    nullptr, adj3, 64, 64,  256, 1);    // adj3 = s2^T@(adj2@s2)
    sg(x3,   W3,   nullptr, tmpA, 64, 128, 128, 0);
    sg(adj3, tmpA, b3,      xg3,  64, 128, 64,  0);    // gcn3
    sg(xg3,  Wd31, nullptr, tmpA, 64, 128, 128, 0);
    sg(adj3, tmpA, bd31,    z3,   64, 128, 64,  0);    // z3
    sg(xg3,  Wd32, nullptr, tmpA, 64, 1,   128, 0);
    sg(adj3, tmpA, bd32,    s3,   64, 1,   64,  0);
    k_softmax_rows<<<64, 256, 0, stream>>>(s3, 1);     // s3 (==1.0)
    sg(s3,   z3,   nullptr, xf,   1, 128, 64, 1);      // xf = s3^T@z3
    k_final<<<1, 64, 0, stream>>>(xf, Wfc, bfc, out);
}
